// PPFOTDetector_62491774157235
// MI455X (gfx1250) — matrix operand order repaired and checked by reading, not compiled, not run on hardware
//
#include <hip/hip_runtime.h>
#include <stdint.h>

// ---------------- problem constants ----------------
#define BATCH 16
#define NS 1024
#define NT 1024
#define DDIM 256
#define REG_INV 10.0f          // 1/0.1
#define KSEL 209715u           // int(NS*NT*0.2)
#define HBINS 65536
#define MU (1.0f / 1024.0f)
#define NU (1.0f / 1024.0f)

typedef __attribute__((ext_vector_type(16))) __bf16 v16bf;
typedef __attribute__((ext_vector_type(8)))  __bf16 v8bf;
typedef __attribute__((ext_vector_type(8)))  float  v8f;

// ---------------- bf16 helpers (manual RNE, bit-exact control) ----------------
__device__ __forceinline__ unsigned short f2bf_bits(float f) {
    uint32_t u = __builtin_bit_cast(uint32_t, f);
    uint32_t r = u + 0x7FFFu + ((u >> 16) & 1u);   // round-to-nearest-even
    return (unsigned short)(r >> 16);
}
__device__ __forceinline__ float bfbits2f(uint32_t hbits) {
    return __builtin_bit_cast(float, hbits << 16);
}

// 16-byte vector load of 8 packed bf16
__device__ __forceinline__ v8bf ld8bf(const uint16_t* p) {
    return __builtin_bit_cast(v8bf, *(const uint4*)p);
}
__device__ __forceinline__ v16bf cat16(v8bf a, v8bf b) {
    return __builtin_shufflevector(a, b, 0, 1, 2, 3, 4, 5, 6, 7,
                                          8, 9, 10, 11, 12, 13, 14, 15);
}

// ---------------- per-row sum of squares (wave32 per row) --------------------
__global__ __launch_bounds__(256) void sumsq_kernel(const float* __restrict__ x,
                                                    float* __restrict__ out,
                                                    int nrows) {
    int row  = blockIdx.x * 8 + (threadIdx.x >> 5);
    int lane = threadIdx.x & 31;
    if (row >= nrows) return;
    const float* p = x + (size_t)row * DDIM;
    float s = 0.0f;
    for (int i = lane; i < DDIM; i += 32) { float v = p[i]; s += v * v; }
    for (int off = 16; off; off >>= 1) s += __shfl_xor(s, off, 32);
    if (lane == 0) out[row] = s;
}

// ---------------- f32 -> (bf16 hi, bf16 lo residual), packed 2/dword ---------
__global__ __launch_bounds__(256) void split_bf16_kernel(const float* __restrict__ x,
                                                         uint32_t* __restrict__ hi,
                                                         uint32_t* __restrict__ lo,
                                                         size_t n2) {
    size_t i  = (size_t)blockIdx.x * blockDim.x + threadIdx.x;
    size_t st = (size_t)gridDim.x * blockDim.x;
    for (; i < n2; i += st) {
        float2 f = ((const float2*)x)[i];
        uint32_t h0 = f2bf_bits(f.x), h1 = f2bf_bits(f.y);
        float r0 = f.x - bfbits2f(h0), r1 = f.y - bfbits2f(h1);
        uint32_t l0 = f2bf_bits(r0), l1 = f2bf_bits(r1);
        hi[i] = h0 | (h1 << 16);
        lo[i] = l0 | (l1 << 16);
    }
}

// ---------------- cost GEMM: bf16x3 WMMA, wave per 16x16 tile, 8 waves/block -
// ISA layouts (CDNA5 ISA §7.12.2):
//   A 16x32: lane holds row (lane&15); VGPRs 0-3 = K base..base+7,
//            VGPRs 4-7 = K base+16..base+23, base = (lane>=16)?8:0
//   B 32x16: lane holds col (lane&15); VGPRs 0-7 = same K as A (base..base+7, base+16..base+23)
//   C/D: lane holds col (lane&15); VGPR j -> row j (+8 for lanes 16..31)
__global__ __launch_bounds__(256) void cost_wmma_kernel(const uint16_t* __restrict__ srcH,
                                                        const uint16_t* __restrict__ srcL,
                                                        const uint16_t* __restrict__ tgtH,
                                                        const uint16_t* __restrict__ tgtL,
                                                        const float* __restrict__ s2,
                                                        const float* __restrict__ t2,
                                                        float* __restrict__ cost) {
    const int b    = blockIdx.z;
    const int wave = threadIdx.x >> 5;
    const int lane = threadIdx.x & 31;          // EXEC all ones around WMMA
    const int sRow = (blockIdx.y * 8 + wave) * 16;
    const int tCol = blockIdx.x * 16;
    const int m     = lane & 15;
    const int abase = (lane & 16) ? 8 : 0;
    const int bbase = (lane & 16) ? 16 : 0;

    const uint16_t* AsH = srcH + ((size_t)b * NS + sRow + m) * DDIM;
    const uint16_t* AsL = srcL + ((size_t)b * NS + sRow + m) * DDIM;
    const uint16_t* BtH = tgtH + ((size_t)b * NT + tCol + m) * DDIM;
    const uint16_t* BtL = tgtL + ((size_t)b * NT + tCol + m) * DDIM;

    v8f acc = {};
#pragma unroll 2
    for (int k0 = 0; k0 < DDIM; k0 += 32) {
        v16bf ah = cat16(ld8bf(AsH + k0 + abase), ld8bf(AsH + k0 + 16 + abase));
        v16bf al = cat16(ld8bf(AsL + k0 + abase), ld8bf(AsL + k0 + 16 + abase));
        v16bf bh = cat16(ld8bf(BtH + k0 + abase), ld8bf(BtH + k0 + 16 + abase));
        v16bf bl = cat16(ld8bf(BtL + k0 + abase), ld8bf(BtL + k0 + 16 + abase));
        // cross ~= ah*bh + ah*bl + al*bh  (bf16x3 ~ fp32 accuracy)
        acc = __builtin_amdgcn_wmma_f32_16x16x32_bf16(false, ah, false, bh, (short)0, acc, false, false);
        acc = __builtin_amdgcn_wmma_f32_16x16x32_bf16(false, ah, false, bl, (short)0, acc, false, false);
        acc = __builtin_amdgcn_wmma_f32_16x16x32_bf16(false, al, false, bh, (short)0, acc, false, false);
    }

    const int n     = tCol + m;
    const float t2n = t2[(size_t)b * NT + n];
    const int rbase = sRow + ((lane & 16) ? 8 : 0);
#pragma unroll
    for (int j = 0; j < 8; ++j) {
        int row = rbase + j;
        float c = s2[(size_t)b * NS + row] + t2n - 2.0f * acc[j];
        c = fmaxf(c, 0.0f);
        cost[((size_t)b * NS + row) * NT + n] = c;
    }
}

// ---------------- generic u32 fill ------------------------------------------
__global__ void fill_u32_kernel(uint32_t* __restrict__ p, uint32_t val, size_t n) {
    size_t i = (size_t)blockIdx.x * blockDim.x + threadIdx.x;
    size_t st = (size_t)gridDim.x * blockDim.x;
    for (; i < n; i += st) p[i] = val;
}

// ---------------- radix select pass 1: hi-16 histogram ----------------------
__global__ __launch_bounds__(256) void hist_hi_kernel(const float* __restrict__ cost,
                                                      uint32_t* __restrict__ hist) {
    const size_t tot = (size_t)BATCH << 20;
    size_t i = (size_t)blockIdx.x * blockDim.x + threadIdx.x;
    size_t st = (size_t)gridDim.x * blockDim.x;
    for (; i < tot; i += st) {
        uint32_t bits = __builtin_bit_cast(uint32_t, cost[i]);
        size_t b = i >> 20;
        atomicAdd(&hist[(b << 16) + (bits >> 16)], 1u);
    }
}

__global__ void scan_hi_kernel(const uint32_t* __restrict__ hist,
                               uint32_t* __restrict__ bucket,
                               uint32_t* __restrict__ remain) {
    int b = threadIdx.x;
    if (b >= BATCH) return;
    const uint32_t* h = hist + ((size_t)b << 16);
    uint64_t cum = 0;
    for (int i = 0; i < HBINS; ++i) {
        uint32_t c = h[i];
        if (cum + c >= (uint64_t)KSEL) { bucket[b] = (uint32_t)i; remain[b] = (uint32_t)(KSEL - cum); return; }
        cum += c;
    }
    bucket[b] = HBINS - 1; remain[b] = 1;
}

// ---------------- radix select pass 2: lo-16 histogram ----------------------
__global__ __launch_bounds__(256) void hist_lo_kernel(const float* __restrict__ cost,
                                                      const uint32_t* __restrict__ bucket,
                                                      uint32_t* __restrict__ hist) {
    const size_t tot = (size_t)BATCH << 20;
    size_t i = (size_t)blockIdx.x * blockDim.x + threadIdx.x;
    size_t st = (size_t)gridDim.x * blockDim.x;
    for (; i < tot; i += st) {
        uint32_t bits = __builtin_bit_cast(uint32_t, cost[i]);
        size_t b = i >> 20;
        if ((bits >> 16) == bucket[b])
            atomicAdd(&hist[(b << 16) + (bits & 0xFFFFu)], 1u);
    }
}

__global__ void scan_lo_kernel(const uint32_t* __restrict__ hist,
                               const uint32_t* __restrict__ bucket,
                               const uint32_t* __restrict__ remain,
                               uint32_t* __restrict__ thr) {
    int b = threadIdx.x;
    if (b >= BATCH) return;
    const uint32_t* h = hist + ((size_t)b << 16);
    uint64_t cum = 0;
    uint32_t need = remain[b];
    uint32_t lo = HBINS - 1;
    for (int i = 0; i < HBINS; ++i) {
        uint32_t c = h[i];
        if (cum + c >= (uint64_t)need) { lo = (uint32_t)i; break; }
        cum += c;
    }
    thr[b] = (bucket[b] << 16) | lo;   // exact bit pattern of k-th smallest cost
}

// ---------------- Gibbs kernel: K = exp(-C/reg) if kept else 1; write K, K^T
__global__ __launch_bounds__(256) void gibbs_kernel(const float* __restrict__ cost,
                                                    const uint32_t* __restrict__ thr,
                                                    uint16_t* __restrict__ Kmat,
                                                    uint16_t* __restrict__ KT) {
    __shared__ uint16_t tile[32][33];
    const int b  = blockIdx.z;
    const int t0 = blockIdx.x * 32;
    const int s0 = blockIdx.y * 32;
    const uint32_t tau = thr[b];
    const int tx = threadIdx.x;   // 0..31
    const int ty = threadIdx.y;   // 0..7
    for (int r = 0; r < 32; r += 8) {
        int s = s0 + ty + r, t = t0 + tx;
        float c = cost[((size_t)b * NS + s) * NT + t];
        uint32_t bits = __builtin_bit_cast(uint32_t, c);
        float kv = (bits <= tau) ? __expf(-c * REG_INV) : 1.0f;  // cost*mask==0 -> exp(0)=1
        uint16_t kb = f2bf_bits(kv);
        Kmat[((size_t)b * NS + s) * NT + t] = kb;
        tile[ty + r][tx] = kb;
    }
    __syncthreads();
    for (int r = 0; r < 32; r += 8) {
        int t = t0 + ty + r, s = s0 + tx;
        KT[((size_t)b * NT + t) * NS + s] = tile[tx][ty + r];
    }
}

// ---------------- GEMV + reciprocal: y[row] = c / dot(Mrow, x) --------------
__global__ __launch_bounds__(256) void gemv_recip_kernel(const uint16_t* __restrict__ M,
                                                         const float* __restrict__ x,
                                                         float* __restrict__ y,
                                                         float cnum) {
    const int b    = blockIdx.y;
    const int row  = blockIdx.x * 8 + (threadIdx.x >> 5);
    const int lane = threadIdx.x & 31;
    const uint32_t* mrow = (const uint32_t*)(M + ((size_t)b * 1024 + row) * 1024); // 512 dwords
    const float* xv = x + (size_t)b * 1024;
    float acc = 0.0f;
#pragma unroll 4
    for (int j = lane; j < 512; j += 32) {
        if (j + 128 < 512) __builtin_prefetch(mrow + j + 128, 0, 0);  // global_prefetch_b8
        uint32_t pk = mrow[j];
        float a0 = bfbits2f(pk & 0xFFFFu);
        float a1 = bfbits2f(pk >> 16);
        acc += a0 * xv[2 * j] + a1 * xv[2 * j + 1];
    }
    for (int off = 16; off; off >>= 1) acc += __shfl_xor(acc, off, 32);
    if (lane == 0) y[(size_t)b * 1024 + row] = cnum / acc;
}

// ---------------- final distance reduction ----------------------------------
__global__ __launch_bounds__(256) void distance_kernel(const float* __restrict__ cost,
                                                       const uint32_t* __restrict__ thr,
                                                       const float* __restrict__ u,
                                                       const float* __restrict__ v,
                                                       float* __restrict__ out) {
    const int b = blockIdx.y;
    const uint32_t tau = thr[b];
    const float* cb = cost + ((size_t)b << 20);
    const float* ub = u + (size_t)b * NS;
    const float* vb = v + (size_t)b * NT;
    float s = 0.0f;
    size_t base = (size_t)blockIdx.x * 4096 + threadIdx.x;
#pragma unroll
    for (int j = 0; j < 16; ++j) {
        size_t e = base + (size_t)j * 256;
        float c = cb[e];
        if (__builtin_bit_cast(uint32_t, c) <= tau) {
            int si = (int)(e >> 10), ti = (int)(e & 1023);
            s += ub[si] * __expf(-c * REG_INV) * vb[ti] * c;   // masked-out: cost term 0
        }
    }
    __shared__ float red[256];
    red[threadIdx.x] = s;
    __syncthreads();
    for (int o = 128; o; o >>= 1) {
        if ((int)threadIdx.x < o) red[threadIdx.x] += red[threadIdx.x + o];
        __syncthreads();
    }
    if (threadIdx.x == 0) atomicAdd(&out[b], red[0]);
}

// ---------------- host orchestration ----------------------------------------
extern "C" void kernel_launch(void* const* d_in, const int* in_sizes, int n_in,
                              void* d_out, int out_size, void* d_ws, size_t ws_size,
                              hipStream_t stream) {
    (void)in_sizes; (void)n_in; (void)out_size; (void)ws_size;
    const float* src = (const float*)d_in[0];   // [B, NS, D] f32
    const float* tgt = (const float*)d_in[1];   // [B, NT, D] f32
    float* out = (float*)d_out;                 // [B] f32

    char* wsB = (char*)d_ws;
    float*    cost   = (float*)wsB;                                  // 64 MB
    uint16_t* Kmat   = (uint16_t*)(wsB + ((size_t)64  << 20));       // 32 MB bf16
    uint16_t* KT     = (uint16_t*)(wsB + ((size_t)96  << 20));       // 32 MB bf16
    // 128..160 MB: bf16 split operands (dead after cost GEMM) ...
    uint16_t* srcH   = (uint16_t*)(wsB + ((size_t)128 << 20));       // 8 MB
    uint16_t* srcL   = (uint16_t*)(wsB + ((size_t)136 << 20));       // 8 MB
    uint16_t* tgtH   = (uint16_t*)(wsB + ((size_t)144 << 20));       // 8 MB
    uint16_t* tgtL   = (uint16_t*)(wsB + ((size_t)152 << 20));       // 8 MB
    // ... aliased afterwards by the radix-select histograms
    uint32_t* histH  = (uint32_t*)(wsB + ((size_t)128 << 20));       // 4 MB
    uint32_t* histL  = (uint32_t*)(wsB + ((size_t)132 << 20));       // 4 MB
    float*    s2     = (float*)(wsB + ((size_t)160 << 20));          // 64 KB
    float*    t2     = s2 + BATCH * NS;                              // 64 KB
    uint32_t* bucket = (uint32_t*)(t2 + BATCH * NT);
    uint32_t* remain = bucket + 16;
    uint32_t* thr    = bucket + 32;
    float*    u      = (float*)(bucket + 64);                        // 64 KB
    float*    v      = u + BATCH * NS;                               // 64 KB

    // 1) row norms + bf16 hi/lo splits (conversion done once, not 64x per elem)
    sumsq_kernel<<<dim3((BATCH * NS) / 8), dim3(256), 0, stream>>>(src, s2, BATCH * NS);
    sumsq_kernel<<<dim3((BATCH * NT) / 8), dim3(256), 0, stream>>>(tgt, t2, BATCH * NT);
    split_bf16_kernel<<<dim3(2048), dim3(256), 0, stream>>>(src, (uint32_t*)srcH, (uint32_t*)srcL,
                                                            (size_t)BATCH * NS * DDIM / 2);
    split_bf16_kernel<<<dim3(2048), dim3(256), 0, stream>>>(tgt, (uint32_t*)tgtH, (uint32_t*)tgtL,
                                                            (size_t)BATCH * NT * DDIM / 2);

    // 2) cost via bf16x3 WMMA (wave per 16x16 tile, 8 waves/block)
    cost_wmma_kernel<<<dim3(NT / 16, NS / 16 / 8, BATCH), dim3(256), 0, stream>>>(
        srcH, srcL, tgtH, tgtL, s2, t2, cost);

    // 3) exact top-k threshold via 2-pass radix select on float bits
    fill_u32_kernel<<<dim3(1024), dim3(256), 0, stream>>>(histH, 0u, (size_t)BATCH * HBINS);
    hist_hi_kernel<<<dim3(4096), dim3(256), 0, stream>>>(cost, histH);
    scan_hi_kernel<<<dim3(1), dim3(32), 0, stream>>>(histH, bucket, remain);
    fill_u32_kernel<<<dim3(1024), dim3(256), 0, stream>>>(histL, 0u, (size_t)BATCH * HBINS);
    hist_lo_kernel<<<dim3(4096), dim3(256), 0, stream>>>(cost, bucket, histL);
    scan_lo_kernel<<<dim3(1), dim3(32), 0, stream>>>(histL, bucket, remain, thr);

    // 4) Gibbs kernel K and K^T (bf16; both live entirely in L2 for the loop)
    gibbs_kernel<<<dim3(NT / 32, NS / 32, BATCH), dim3(32, 8), 0, stream>>>(cost, thr, Kmat, KT);

    // 5) init u = v = 1, out = 0
    const uint32_t ONE = __builtin_bit_cast(uint32_t, 1.0f);
    fill_u32_kernel<<<dim3(64), dim3(256), 0, stream>>>((uint32_t*)u, ONE, (size_t)BATCH * NS);
    fill_u32_kernel<<<dim3(64), dim3(256), 0, stream>>>((uint32_t*)v, ONE, (size_t)BATCH * NT);
    fill_u32_kernel<<<dim3(1), dim3(32), 0, stream>>>((uint32_t*)out, 0u, (size_t)BATCH);

    // 6) 100 Sinkhorn iterations: u = mu/(K v); v = nu/(K^T u)
    for (int it = 0; it < 100; ++it) {
        gemv_recip_kernel<<<dim3(NS / 8, BATCH), dim3(256), 0, stream>>>(Kmat, v, u, MU);
        gemv_recip_kernel<<<dim3(NT / 8, BATCH), dim3(256), 0, stream>>>(KT, u, v, NU);
    }

    // 7) distance[b] = sum over kept entries of u_s * exp(-c/reg) * v_t * c
    distance_kernel<<<dim3(256, BATCH), dim3(256), 0, stream>>>(cost, thr, u, v, out);
}
